// ParallelFLAAttention_15350213115966
// MI455X (gfx1250) — matrix operand order repaired and checked by reading, not compiled, not run on hardware
//
#include <hip/hip_runtime.h>
#include <hip/hip_bf16.h>
#include <math.h>

// Problem constants (from reference)
#define B_    2
#define N_    2048
#define HID_  2048
#define H_    16
#define KVH_  4
#define D_    128
#define GROUPS_ (H_ / KVH_)
#define NEGINF (-1e30f)

typedef __bf16 bf16_t;
typedef __attribute__((ext_vector_type(16))) __bf16 v16bf;
typedef __attribute__((ext_vector_type(8)))  float  v8f;

__device__ __forceinline__ v8f wmma_bf16(v16bf a, v16bf b, v8f c) {
  // D = A(16x32 bf16) * B(32x16 bf16) + C(16x16 f32)
  return __builtin_amdgcn_wmma_f32_16x16x32_bf16(false, a, false, b, (short)0, c,
                                                 false, false);
}

// K-dim permutation within a 32-group so each lane's A-fragment elements are
// contiguous: positions [0..7]=k0-7, [8..15]=k16-23, [16..23]=k8-15,
// [24..31]=k24-31.  Lane half `hi` then reads 16 contiguous bf16 (32 bytes).
__device__ __forceinline__ int kperm(int k) {
  return (k & 7) | (((k >> 4) & 1) << 3) | (((k >> 3) & 1) << 4);
}

// Pack 4 floats -> 4 bf16 -> single 8-byte LDS store (dst must be 8B aligned).
__device__ __forceinline__ void store4bf(bf16_t* dst, float a, float b, float c,
                                         float d) {
  union { bf16_t h[4]; uint2 u; } pk;
  pk.h[0] = (bf16_t)a; pk.h[1] = (bf16_t)b; pk.h[2] = (bf16_t)c; pk.h[3] = (bf16_t)d;
  *(uint2*)dst = pk.u;
}

__device__ __forceinline__ float fget(const float4& f, int j) {
  return j == 0 ? f.x : j == 1 ? f.y : j == 2 ? f.z : f.w;
}

// ---------------------------------------------------------------------------
// Generic tiled GEMM:  C[M,N] = A[M,K] * B[K,N]  (+ addScale * addend)
// fp32 in/out, bf16 WMMA compute. Tile 128x128, BK=32. 256 threads = 8 waves,
// each wave computes a 32x64 sub-tile (2x4 WMMA accumulators).
// Double-buffered LDS: one barrier per K-step; stage tile kt+1 while the
// WMMAs consume tile kt.  A staged with permuted K; B staged transposed so
// every fragment load is a contiguous 32B vector LDS load.
// ---------------------------------------------------------------------------
template <bool HAS_ADD>
__global__ __launch_bounds__(256)
void gemm_bf16_wmma(const float* __restrict__ A, const float* __restrict__ Bm,
                    float* __restrict__ C, int M, int Nc, int K,
                    const float* __restrict__ addend, float addScale) {
  __shared__ __align__(32) bf16_t As[2][128 * 32];  // [m][kperm]
  __shared__ __align__(32) bf16_t Bt[2][128 * 32];  // [n][kperm] (transposed)

  const int tid   = threadIdx.x;
  const int lane  = tid & 31;
  const int wave  = tid >> 5;
  const int wm    = (wave >> 1) * 32;
  const int wn    = (wave & 1) * 64;
  const int row0  = blockIdx.x * 128;
  const int col0  = blockIdx.y * 128;
  const int nlane = lane & 15;
  const int hi    = lane >> 4;
  // B staging decomposition: 8 k-blocks x 32 n-blocks of 4x4
  const int kb = tid >> 5;
  const int nb = tid & 31;

  v8f acc[2][4];
#pragma unroll
  for (int i = 0; i < 2; ++i)
#pragma unroll
    for (int j = 0; j < 4; ++j)
#pragma unroll
      for (int e = 0; e < 8; ++e) acc[i][j][e] = 0.f;

  auto stage = [&](int buf, int k0) {
    // A tile 128x32: 1024 float4 quads, permuted-K packed stores.
#pragma unroll
    for (int t4 = 0; t4 < 4; ++t4) {
      int it = tid + t4 * 256;
      int r  = it >> 3;
      int c4 = (it & 7) * 4;
      float4 x = *(const float4*)&A[(size_t)(row0 + r) * K + k0 + c4];
      store4bf(&As[buf][r * 32 + kperm(c4)], x.x, x.y, x.z, x.w);
    }
    // B tile 32x128 transposed via 4x4 register blocks.
    float4 x[4];
#pragma unroll
    for (int i = 0; i < 4; ++i)
      x[i] = *(const float4*)&Bm[(size_t)(k0 + kb * 4 + i) * Nc + col0 + nb * 4];
#pragma unroll
    for (int j = 0; j < 4; ++j)
      store4bf(&Bt[buf][(nb * 4 + j) * 32 + kperm(kb * 4)], fget(x[0], j), fget(x[1], j),
               fget(x[2], j), fget(x[3], j));
  };

  const int KT = K >> 5;
  stage(0, 0);
  for (int kt = 0; kt < KT; ++kt) {
    const int cur = kt & 1;
    __syncthreads();
    if (kt + 1 < KT) {
      __builtin_prefetch(&A[(size_t)(row0 + (tid & 127)) * K + (kt + 2) * 32], 0, 1);
      __builtin_prefetch(&Bm[(size_t)((kt + 2) * 32 + (tid & 31)) * Nc + col0], 0, 1);
      stage(cur ^ 1, (kt + 1) * 32);
    }
    v16bf af[2];
#pragma unroll
    for (int i = 0; i < 2; ++i)
      af[i] = *(const v16bf*)&As[cur][(wm + i * 16 + nlane) * 32 + hi * 16];
    v16bf bfg[4];
#pragma unroll
    for (int j = 0; j < 4; ++j)
      bfg[j] = *(const v16bf*)&Bt[cur][(wn + j * 16 + nlane) * 32 + hi * 16];
#pragma unroll
    for (int i = 0; i < 2; ++i)
#pragma unroll
      for (int j = 0; j < 4; ++j)
        acc[i][j] = wmma_bf16(af[i], bfg[j], acc[i][j]);
  }

  // Epilogue: C/D layout — VGPR e is row e (lanes 0-15) or row 8+e (16-31).
#pragma unroll
  for (int i = 0; i < 2; ++i)
#pragma unroll
    for (int j = 0; j < 4; ++j) {
      int r0 = row0 + wm + i * 16 + (hi ? 8 : 0);
      int c  = col0 + wn + j * 16 + nlane;
#pragma unroll
      for (int e = 0; e < 8; ++e) {
        size_t idx = (size_t)(r0 + e) * Nc + c;
        float val  = acc[i][j][e];
        if (HAS_ADD) val += addScale * addend[idx];
        C[idx] = val;
      }
    }
}

// ---------------------------------------------------------------------------
// Causal flash attention with GQA, bf16 WMMA, online softmax.
// Grid: (N/128, H, B). Block: 256 threads = 8 waves; wave w owns query rows
// [q0+16w, q0+16w+16). KV tile = 64 keys.
// sK kept [key][dperm] (S B-operand is one 32B vector load); sV stored transposed
// [d][keyperm] (PV B-operand is one 32B vector load); P scratch uses permuted-K
// layout so the A-operand reload is one 32B vector load.
// ---------------------------------------------------------------------------
__global__ __launch_bounds__(256)
void flash_attn_wmma(const float* __restrict__ q, const float* __restrict__ k,
                     const float* __restrict__ v, float* __restrict__ o) {
  __shared__ __align__(32) bf16_t sK[64 * 128];    // [key][dperm]
  __shared__ __align__(32) bf16_t sVt[128 * 64];   // [d][keyperm]
  __shared__ __align__(32) bf16_t sP[8 * 16 * 64]; // per-wave, [m][keyperm]

  const int tid   = threadIdx.x;
  const int lane  = tid & 31;
  const int wave  = tid >> 5;
  const int nlane = lane & 15;
  const int hi    = lane >> 4;
  const int qt = blockIdx.x, h = blockIdx.y, b = blockIdx.z;
  const int kh = h / GROUPS_;
  const int q0 = qt * 128;
  const int qrow = q0 + wave * 16;
  const float scale = 0.08838834764831845f;  // 1/sqrt(128)

  // Q fragments (A operand, 4 chunks of 32 over D) straight from global.
  v16bf qf[4];
  {
    const float* qp = q + (size_t)(b * N_ + qrow + nlane) * (H_ * D_) + h * D_;
#pragma unroll
    for (int c = 0; c < 4; ++c) {
      float4 a0 = *(const float4*)&qp[c * 32 + hi * 8];
      float4 a1 = *(const float4*)&qp[c * 32 + hi * 8 + 4];
      float4 b0 = *(const float4*)&qp[c * 32 + 16 + hi * 8];
      float4 b1 = *(const float4*)&qp[c * 32 + 16 + hi * 8 + 4];
      v16bf f;
#pragma unroll
      for (int e = 0; e < 4; ++e) {
        f[e]      = (bf16_t)fget(a0, e);
        f[4 + e]  = (bf16_t)fget(a1, e);
        f[8 + e]  = (bf16_t)fget(b0, e);
        f[12 + e] = (bf16_t)fget(b1, e);
      }
      qf[c] = f;
    }
  }

  v8f oacc[8];
#pragma unroll
  for (int j = 0; j < 8; ++j)
#pragma unroll
    for (int e = 0; e < 8; ++e) oacc[j][e] = 0.f;
  float mrow[8], lrow[8];
#pragma unroll
  for (int e = 0; e < 8; ++e) { mrow[e] = NEGINF; lrow[e] = 0.f; }

  const int ntiles = q0 / 64 + 2;  // causal: keys up to end of this Q tile
  for (int jt = 0; jt < ntiles; ++jt) {
    const int kv0 = jt * 64;
    // Stage K tile (64x128): 2048 quads, packed 8B stores.
#pragma unroll
    for (int t8 = 0; t8 < 8; ++t8) {
      int it = tid + t8 * 256;
      int r  = it >> 5;
      int c4 = (it & 31) * 4;
      float4 x = *(const float4*)&k[(size_t)(b * N_ + kv0 + r) * (KVH_ * D_) +
                                    kh * D_ + c4];
      store4bf(&sK[r * 128 + (c4 & ~31) + kperm(c4 & 31)], x.x, x.y, x.z, x.w);
    }
    // Stage V tile transposed (sVt[d][key]) via 4x4 blocks: 16 kb x 32 db.
#pragma unroll
    for (int t2 = 0; t2 < 2; ++t2) {
      int blkid = tid + t2 * 256;      // 0..511
      int kb2 = blkid >> 5;            // keys kb2*4..+3
      int db  = blkid & 31;            // d db*4..+3
      float4 x[4];
#pragma unroll
      for (int i = 0; i < 4; ++i)
        x[i] = *(const float4*)&v[(size_t)(b * N_ + kv0 + kb2 * 4 + i) *
                                      (KVH_ * D_) + kh * D_ + db * 4];
#pragma unroll
      for (int j = 0; j < 4; ++j)
        store4bf(&sVt[(db * 4 + j) * 64 + ((kb2 * 4) & 32) + kperm((kb2 * 4) & 31)], fget(x[0], j), fget(x[1], j),
                 fget(x[2], j), fget(x[3], j));
    }
    __syncthreads();

    // S = Q * K^T : 4 n-tiles x 4 d-chunks
    v8f s[4];
#pragma unroll
    for (int j = 0; j < 4; ++j)
#pragma unroll
      for (int e = 0; e < 8; ++e) s[j][e] = 0.f;
#pragma unroll
    for (int j = 0; j < 4; ++j)
#pragma unroll
      for (int c = 0; c < 4; ++c) {
        v16bf f = *(const v16bf*)&sK[(j * 16 + nlane) * 128 + c * 32 + hi * 16];
        s[j] = wmma_bf16(qf[c], f, s[j]);
      }

    // Scale (+ causal mask only on tiles that can touch the diagonal)
    float pm[4][8];
#pragma unroll
    for (int j = 0; j < 4; ++j)
#pragma unroll
      for (int e = 0; e < 8; ++e) pm[j][e] = s[j][e] * scale;
    if (kv0 + 63 > qrow) {  // wave-uniform
#pragma unroll
      for (int j = 0; j < 4; ++j)
#pragma unroll
        for (int e = 0; e < 8; ++e) {
          int kidx = kv0 + j * 16 + nlane;
          int qidx = qrow + hi * 8 + e;
          if (kidx > qidx) pm[j][e] = NEGINF;
        }
    }
    // Row max across the 16 lanes holding each row
    float rmax[8];
#pragma unroll
    for (int e = 0; e < 8; ++e) {
      float m = pm[0][e];
#pragma unroll
      for (int j = 1; j < 4; ++j) m = fmaxf(m, pm[j][e]);
#pragma unroll
      for (int d = 1; d < 16; d <<= 1) m = fmaxf(m, __shfl_xor(m, d, 32));
      rmax[e] = m;
    }
    float alpha[8];
#pragma unroll
    for (int e = 0; e < 8; ++e) {
      float mnew = fmaxf(mrow[e], rmax[e]);
      alpha[e]   = __expf(mrow[e] - mnew);
      mrow[e]    = mnew;
    }
    float rsum[8];
#pragma unroll
    for (int e = 0; e < 8; ++e) rsum[e] = 0.f;
#pragma unroll
    for (int j = 0; j < 4; ++j)
#pragma unroll
      for (int e = 0; e < 8; ++e) {
        float pv = __expf(pm[j][e] - mrow[e]);
        pm[j][e] = pv;
        rsum[e] += pv;
      }
#pragma unroll
    for (int e = 0; e < 8; ++e) {
#pragma unroll
      for (int d = 1; d < 16; d <<= 1) rsum[e] += __shfl_xor(rsum[e], d, 32);
      lrow[e] = lrow[e] * alpha[e] + rsum[e];
    }
#pragma unroll
    for (int j = 0; j < 8; ++j)
#pragma unroll
      for (int e = 0; e < 8; ++e) oacc[j][e] *= alpha[e];

    // P -> per-wave LDS in permuted-K A layout
    bf16_t* pw = sP + wave * (16 * 64);
#pragma unroll
    for (int j = 0; j < 4; ++j) {
      int pos = ((j >> 1) << 5) + kperm(((j & 1) << 4) + nlane);
#pragma unroll
      for (int e = 0; e < 8; ++e)
        pw[(hi * 8 + e) * 64 + pos] = (bf16_t)pm[j][e];
    }
    __syncthreads();

    // O += P * V : 2 key-chunks x 8 d-tiles
    v16bf pf[2];
#pragma unroll
    for (int c = 0; c < 2; ++c)
      pf[c] = *(const v16bf*)&pw[nlane * 64 + c * 32 + hi * 16];
#pragma unroll
    for (int jn = 0; jn < 8; ++jn)
#pragma unroll
      for (int c = 0; c < 2; ++c) {
        v16bf f = *(const v16bf*)&sVt[(jn * 16 + nlane) * 64 + c * 32 + hi * 16];
        oacc[jn] = wmma_bf16(pf[c], f, oacc[jn]);
      }
    __syncthreads();
  }

  // Normalize and store O (base buffer, layout [b*N+n][h*D+d])
#pragma unroll
  for (int e = 0; e < 8; ++e) {
    float inv = 1.f / lrow[e];
    size_t row = (size_t)(b * N_ + qrow + hi * 8 + e) * (H_ * D_);
#pragma unroll
    for (int jn = 0; jn < 8; ++jn)
      o[row + h * D_ + jn * 16 + nlane] = oacc[jn][e] * inv;
  }
}

// ---------------------------------------------------------------------------
// Gated linear attention recurrence (sequential over n). One block per (b,h).
// 256 threads: thread (dhalf, dv) owns S[d, dv] for d in [dhalf*64, +64).
// ---------------------------------------------------------------------------
__global__ __launch_bounds__(256)
void gla_recurrence(const float* __restrict__ q, const float* __restrict__ k,
                    const float* __restrict__ v, float* __restrict__ fla) {
  const int b = blockIdx.x / H_;
  const int h = blockIdx.x % H_;
  const int kh = h / GROUPS_;
  const int t  = threadIdx.x;
  const int dv = t & 127;
  const int dhalf = t >> 7;

  float S[64];
#pragma unroll
  for (int i = 0; i < 64; ++i) S[i] = 0.f;

  __shared__ float sk[128], sv_[128], sq[128], sg[128];
  __shared__ float partial[256];
  const float scale = 0.08838834764831845f;

  for (int n = 0; n < N_; ++n) {
    if (t < 128) {
      size_t kvrow = (size_t)(b * N_ + n) * (KVH_ * D_) + kh * D_ + t;
      float kt = k[kvrow];
      sk[t] = kt;
      sg[t] = 1.f / (1.f + __expf(-kt));
      sv_[t] = v[kvrow];
      sq[t] = q[(size_t)(b * N_ + n) * (H_ * D_) + h * D_ + t];
    }
    __syncthreads();
    float vv  = sv_[dv];
    float acc = 0.f;
#pragma unroll 8
    for (int i = 0; i < 64; ++i) {
      int d   = dhalf * 64 + i;
      float s = S[i] * sg[d] + sk[d] * vv;
      S[i]    = s;
      acc    += sq[d] * s;
    }
    partial[t] = acc;
    __syncthreads();
    if (t < 128)
      fla[(size_t)(b * N_ + n) * (H_ * D_) + h * D_ + t] =
          scale * (partial[t] + partial[t + 128]);
  }
}

// ---------------------------------------------------------------------------
extern "C" void kernel_launch(void* const* d_in, const int* in_sizes, int n_in,
                              void* d_out, int out_size, void* d_ws, size_t ws_size,
                              hipStream_t stream) {
  (void)in_sizes; (void)n_in; (void)out_size; (void)ws_size;
  const float* hs = (const float*)d_in[0];
  const float* Wq = (const float*)d_in[1];
  const float* Wk = (const float*)d_in[2];
  const float* Wv = (const float*)d_in[3];
  const float* Wo = (const float*)d_in[4];
  float* out = (float*)d_out;

  const int M = B_ * N_;  // 4096
  float* qb   = (float*)d_ws;                       // M x H*D
  float* kb   = qb   + (size_t)M * H_ * D_;         // M x KVH*D
  float* vb   = kb   + (size_t)M * KVH_ * D_;       // M x KVH*D
  float* base = vb   + (size_t)M * KVH_ * D_;       // M x H*D
  float* flab = base + (size_t)M * H_ * D_;         // M x H*D

  dim3 blk(256);
  gemm_bf16_wmma<false><<<dim3(M / 128, (H_ * D_) / 128), blk, 0, stream>>>(
      hs, Wq, qb, M, H_ * D_, HID_, nullptr, 0.f);
  gemm_bf16_wmma<false><<<dim3(M / 128, (KVH_ * D_) / 128), blk, 0, stream>>>(
      hs, Wk, kb, M, KVH_ * D_, HID_, nullptr, 0.f);
  gemm_bf16_wmma<false><<<dim3(M / 128, (KVH_ * D_) / 128), blk, 0, stream>>>(
      hs, Wv, vb, M, KVH_ * D_, HID_, nullptr, 0.f);
  flash_attn_wmma<<<dim3(N_ / 128, H_, B_), blk, 0, stream>>>(qb, kb, vb, base);
  gla_recurrence<<<dim3(B_ * H_), blk, 0, stream>>>(qb, kb, vb, flab);
  gemm_bf16_wmma<true><<<dim3(M / 128, HID_ / 128), blk, 0, stream>>>(
      base, Wo, out, M, HID_, H_ * D_, flab, 0.01f);
}
